// NHP_53154515255447
// MI455X (gfx1250) — compile-verified, matrix operand order checked by reading, not run on hardware
//
#include <hip/hip_runtime.h>

// ---------------- CDNA5 WMMA / TDM types ----------------
typedef __attribute__((ext_vector_type(16))) __bf16 v16bf;
typedef __attribute__((ext_vector_type(8)))  float  v8f;
typedef __attribute__((ext_vector_type(4)))  unsigned int v4u;
typedef __attribute__((ext_vector_type(8)))  int v8i;
typedef __attribute__((ext_vector_type(4)))  int v4i;

#define HAS_TDM __has_builtin(__builtin_amdgcn_tensor_load_to_lds)

#define NHP_B    64
#define NHP_L    512
#define NHP_E    128
#define NHP_H    256
#define NHP_G    1792         // 7*H
#define NHP_NT   20
#define NHP_S    5
#define NWG      8
#define HSLICE   32           // H / NWG
#define NCT      14           // (7 gates * 32 cols) / 16
#define NKT      12           // 384 / 32
#define NRT      4            // 64 / 16

// LDS layout (bytes)
#define OFF_W    0                         // 14*12*512*2 = 172032
#define OFF_A    172032                    // 48*512*2    = 49152 (tiles ordered kt*4+rt; aliased by HS f32)
#define OFF_G    221184                    // 64*7*32*4   = 57344
#define OFF_C    278528                    // 64*32*4     = 8192
#define OFF_CB   286720                    // 64*32*4     = 8192
#define OFF_ACC  294912                    // 2*64*4      = 512
#define LDS_BYTES 295424

// workspace layout (bytes)
#define WS_X     0                         // 512 steps * 16KB  (bf16 x-tiles, WMMA frag order)
#define WS_H     8388608                   // 32 tiles * 1KB    (bf16 h-tiles, WMMA frag order)
#define WS_PSUM  8421376                   // 2*6*8*64*20*4 = 491520
#define WS_BAR   8912896                   // 2 * u32

__device__ __forceinline__ unsigned short f2bf(float f) {
    union { float f; unsigned u; } v; v.f = f;
    unsigned r = v.u + 0x7FFFu + ((v.u >> 16) & 1u);
    return (unsigned short)(r >> 16);
}
// K index (within a 32-wide K tile) held by element ii (0..15) of a lane in half h (0/1)
__device__ __forceinline__ int kof(int half, int ii) {
    int p = ii >> 1, odd = ii & 1;
    int base = (p < 4) ? (p << 1) : (16 + ((p - 4) << 1));
    return base + (half << 3) + odd;
}
__device__ __forceinline__ float sigm(float x)  { return 1.f / (1.f + __expf(-x)); }
__device__ __forceinline__ float splus(float x) { return (x > 20.f) ? x : log1pf(__expf(x)); }

// ---- Tensor Data Mover: 1-D contiguous global->LDS copy (n8 x 8-byte elements) ----
__device__ __forceinline__ void tdm_copy_to_lds(unsigned lds_off, const void* gaddr, unsigned n8) {
#if HAS_TDM
    unsigned long long ga = (unsigned long long)gaddr;
    v4u g0;
    g0[0] = 1u;                                              // count=1, user mode
    g0[1] = lds_off;                                         // lds_addr (bytes)
    g0[2] = (unsigned)(ga & 0xFFFFFFFFull);                  // global_addr[31:0]
    g0[3] = ((unsigned)(ga >> 32) & 0x01FFFFFFu)             // global_addr[56:32]
            | 0x80000000u;                                   // type = 2 ("image")
    v8i g1;
    g1[0] = (int)(3u << 16);                                 // data_size = 8B
    g1[1] = (int)((n8 & 0xFFFFu) << 16);                     // tensor_dim0[15:0]
    g1[2] = (int)((n8 >> 16) | (1u << 16));                  // tensor_dim0[31:16] | tensor_dim1=1
    g1[3] = (int)((n8 & 0xFFFFu) << 16);                     // tile_dim0
    g1[4] = 1;                                               // tile_dim1 = 1
    g1[5] = (int)n8;                                         // tensor_dim0_stride
    g1[6] = 0; g1[7] = 0;
    v4i z4 = {0, 0, 0, 0};
    v8i z8 = {0, 0, 0, 0, 0, 0, 0, 0};
    __builtin_amdgcn_tensor_load_to_lds(g0, g1, z4, z4, z8, 0);
#else
    (void)lds_off; (void)gaddr; (void)n8;
#endif
}

__device__ __forceinline__ void grid_barrier(unsigned* cnt, unsigned* gen, unsigned& lgen) {
    __threadfence();
    __syncthreads();
    if (threadIdx.x == 0) {
        __builtin_amdgcn_s_cluster_barrier();   // real sync when cluster-launched (NOP otherwise)
        unsigned target = ++lgen;
        unsigned arrived = atomicAdd(cnt, 1u) + 1u;
        if (arrived == target * (unsigned)NWG) {
            __threadfence();
            atomicExch(gen, target);
        } else {
            while (atomicAdd(gen, 0u) < target) __builtin_amdgcn_s_sleep(2);
        }
    }
    __syncthreads();
}

// ------- prologue: embed gather -> bf16 in WMMA-tile order, zero h-tiles & barrier -------
__global__ void nhp_prologue(const int* __restrict__ type_seq,
                             const float* __restrict__ embed,
                             unsigned short* __restrict__ Xts,
                             unsigned short* __restrict__ Hts,
                             unsigned* __restrict__ bar) {
    long idx = (long)blockIdx.x * 256 + threadIdx.x;     // over 512*16*512 elements
    if (idx < (long)NHP_L * 16 * 512) {
        int e    = (int)(idx & 511);
        int tile = (int)((idx >> 9) & 15);               // kt*4 + rt, kt in 0..3
        int t    = (int)(idx >> 13);
        int kt = tile >> 2, rt = tile & 3;
        int ln = e >> 4, ii = e & 15;
        int m = rt * 16 + (ln & 15);
        int k = kt * 32 + kof(ln >> 4, ii);
        int ty = type_seq[m * (NHP_L + 1) + t];
        Xts[idx] = f2bf(embed[(long)ty * NHP_E + k]);
    }
    if (idx < 32 * 512) Hts[idx] = 0;                    // h0 = 0 (tiled layout)
    if (idx < 4) bar[idx] = 0;
}

// ------------- persistent fused scan kernel (one workgroup per WGP) -------------
__global__ __launch_bounds__(256, 1)
void nhp_scan(const int* __restrict__ type_seq, const float* __restrict__ time_seq,
              const float* __restrict__ u, const float* __restrict__ W_gates,
              const float* __restrict__ b_gates, const float* __restrict__ W_int,
              const unsigned short* __restrict__ Xts, unsigned short* __restrict__ Hts,
              float* __restrict__ psum, unsigned* __restrict__ bar,
              float* __restrict__ out) {
    extern __shared__ char smem[];
    unsigned short* Wl = (unsigned short*)(smem + OFF_W);
    unsigned short* Al = (unsigned short*)(smem + OFF_A);
    float* HSl  = (float*)(smem + OFF_A);     // alias: h_s staging after A consumed
    float* Gl   = (float*)(smem + OFF_G);
    float* Cl   = (float*)(smem + OFF_C);
    float* CBl  = (float*)(smem + OFF_CB);
    float* accI = (float*)(smem + OFF_ACC);
    float* accL = accI + NHP_B;

    const int wg   = blockIdx.x;
    const int j0   = wg * HSLICE;
    const int lane = threadIdx.x & 31;
    const int wave = threadIdx.x >> 5;
    const int nA   = lane & 15;
    const int half = lane >> 4;
    const int rt   = wave >> 1;               // one row-tile per wave
    const int ct0  = (wave & 1) * 7;          // 7 consecutive col-tiles per wave

    // ---- load this WGP's recurrent-weight slice into LDS (bf16), once ----
    for (int i = threadIdx.x; i < NCT * NKT * 512; i += 256) {
        int tile = i >> 9, e = i & 511;
        int kt = tile % NKT, ct = tile / NKT;
        int ln = e >> 4, ii = e & 15;
        int q = ct >> 1;
        int jj = ((ct & 1) << 4) + (ln & 15);
        int col = q * NHP_H + j0 + jj;
        int k = kt * 32 + kof(ln >> 4, ii);
        Wl[i] = f2bf(W_gates[(long)k * NHP_G + col]);
    }
    for (int i = threadIdx.x; i < NHP_B * HSLICE; i += 256) { Cl[i] = 0.f; CBl[i] = 0.f; }
    if (threadIdx.x < NHP_B) { accI[threadIdx.x] = 0.f; accL[threadIdx.x] = 0.f; }
    unsigned* g_cnt = bar, * g_gen = bar + 1;
    unsigned lgen = 0;

    for (int t = 0; t < NHP_L; ++t) {
        grid_barrier(g_cnt, g_gen, lgen);           // h-tiles from step t-1 visible everywhere

        // ---- stage A = [x_t | h] : both sources are pre-swizzled WMMA tiles ----
#if HAS_TDM
        if (threadIdx.x == 0) {
            tdm_copy_to_lds(OFF_A,         (const char*)Xts + (size_t)t * 16384, 2048); // 16KB x-tiles
            tdm_copy_to_lds(OFF_A + 16384, (const char*)Hts,                     4096); // 32KB h-tiles
            __builtin_amdgcn_s_wait_tensorcnt(0);
        }
        __syncthreads();
#else
        {
            const uint4* xs = (const uint4*)(Xts + (size_t)t * 8192);
            const uint4* hs4 = (const uint4*)Hts;
            uint4* ad = (uint4*)Al;
            for (int i = threadIdx.x; i < 1024; i += 256) ad[i] = xs[i];
            for (int i = threadIdx.x; i < 2048; i += 256) ad[1024 + i] = hs4[i];
            __syncthreads();
        }
#endif

        // ---- GEMM slice: batch-load 1 A-frag + 7 B-frags, then 7 back-to-back WMMAs ----
        v8f acc[7];
        #pragma unroll
        for (int tt = 0; tt < 7; ++tt)
            #pragma unroll
            for (int r = 0; r < 8; ++r) acc[tt][r] = 0.f;
        #pragma unroll
        for (int kt = 0; kt < NKT; ++kt) {
            v16bf a = *(const v16bf*)(Al + (((kt << 2) + rt) << 9) + (lane << 4));
            v16bf bwf[7];
            #pragma unroll
            for (int tt = 0; tt < 7; ++tt)
                bwf[tt] = *(const v16bf*)(Wl + (((ct0 + tt) * NKT + kt) << 9) + (lane << 4));
            #pragma unroll
            for (int tt = 0; tt < 7; ++tt)
                acc[tt] = __builtin_amdgcn_wmma_f32_16x16x32_bf16(
                    false, a, false, bwf[tt], (short)0, acc[tt], false, false);
        }
        #pragma unroll
        for (int tt = 0; tt < 7; ++tt) {
            int ct = ct0 + tt;
            int q = ct >> 1, jb = (ct & 1) << 4;
            #pragma unroll
            for (int r = 0; r < 8; ++r) {
                int m = rt * 16 + r + (half << 3);
                Gl[(m * 7 + q) * HSLICE + jb + nA] = acc[tt][r];
            }
        }
        __syncthreads();   // Al free from here -> HSl alias usable

        // ---- CT-LSTM elementwise update for this WGP's 32 hidden units ----
        #pragma unroll
        for (int r = 0; r < 8; ++r) {
            int idx = threadIdx.x + (r << 8);        // 0..2047 = (b, jj)
            int b = idx >> 5, jj = idx & 31;
            int j = j0 + jj;
            int gb = (b * 7) * HSLICE + jj;
            float gi = sigm (Gl[gb + 0 * HSLICE] + b_gates[0 * NHP_H + j]);
            float gf = sigm (Gl[gb + 1 * HSLICE] + b_gates[1 * NHP_H + j]);
            float gz = tanhf(Gl[gb + 2 * HSLICE] + b_gates[2 * NHP_H + j]);
            float go = sigm (Gl[gb + 3 * HSLICE] + b_gates[3 * NHP_H + j]);
            float gib= sigm (Gl[gb + 4 * HSLICE] + b_gates[4 * NHP_H + j]);
            float gfb= sigm (Gl[gb + 5 * HSLICE] + b_gates[5 * NHP_H + j]);
            float gd = splus(Gl[gb + 6 * HSLICE] + b_gates[6 * NHP_H + j]);
            float c   = gf * Cl[b * HSLICE + jj] + gi * gz;
            float cbn = gfb * CBl[b * HSLICE + jj] + gib * gz;
            int kst = type_seq[b * (NHP_L + 1) + t + 1];
            float dt = (kst == 0) ? 0.f
                       : (time_seq[b * (NHP_L + 1) + t + 1] - time_seq[b * (NHP_L + 1) + t]);
            float cdn = cbn + (c - cbn) * __expf(-gd * dt);
            float hdn = go * tanhf(cdn);
            Cl[b * HSLICE + jj] = c; CBl[b * HSLICE + jj] = cbn;
            // write new h directly in WMMA-fragment order (this WG's slice == K-tile 4+wg)
            {
                int half_h = (jj >> 3) & 1;
                int base = jj - (half_h << 3);                    // in {0..7} U {16..23}
                int p = (base < 8) ? (base >> 1) : (4 + ((base - 16) >> 1));
                int iih = (p << 1) | (base & 1);
                int lane_h = (b & 15) + (half_h << 4);
                Hts[((wg * 4 + (b >> 4)) << 9) + (lane_h << 4) + iih] = f2bf(hdn);
            }
            HSl[(0 * NHP_B + b) * HSLICE + jj] = hdn;
            #pragma unroll
            for (int s = 0; s < NHP_S; ++s) {
                float uu = u[((long)b * NHP_L + t) * NHP_S + s];
                float cts = cbn + (c - cbn) * __expf(-gd * uu * dt);
                HSl[((1 + s) * NHP_B + b) * HSLICE + jj] = go * tanhf(cts);
            }
        }
        __syncthreads();

        // ---- partial (H-slice) GEMV vs W_int -> L2 scratch (double-buffered) ----
        int buf = t & 1;
        #pragma unroll
        for (int p = 0; p < 5; ++p) {
            int idx = threadIdx.x + (p << 8);
            if (idx < NHP_B * NHP_NT) {
                int b = idx / NHP_NT, k = idx % NHP_NT;
                #pragma unroll
                for (int ch = 0; ch < 6; ++ch) {
                    float s = 0.f;
                    #pragma unroll 8
                    for (int jj = 0; jj < HSLICE; ++jj)
                        s += HSl[(ch * NHP_B + b) * HSLICE + jj] * W_int[(j0 + jj) * NHP_NT + k];
                    psum[((((long)buf * 6 + ch) * NWG + wg) * NHP_B + b) * NHP_NT + k] = s;
                }
            }
        }
        grid_barrier(g_cnt, g_gen, lgen);           // psums of step t complete

        // ---- WGP0 reduces + softplus + loss accumulation (others race ahead) ----
        if (wg == 0 && threadIdx.x < NHP_B) {
            int b = threadIdx.x;
            int kst = type_seq[b * (NHP_L + 1) + t + 1];
            if (kst != 0) {
                float dt = time_seq[b * (NHP_L + 1) + t + 1] - time_seq[b * (NHP_L + 1) + t];
                int col = ((kst < 20) ? kst : 20) - 1;      // take_along_axis clamp semantics
                float ev = 0.f;
                for (int w8 = 0; w8 < NWG; ++w8)
                    ev += psum[((((long)buf * 6 + 0) * NWG + w8) * NHP_B + b) * NHP_NT + col];
                ev = fmaxf(splus(ev), 1e-10f);
                accL[b] += __logf(ev);
                float tot = 0.f;
                for (int s = 0; s < NHP_S; ++s)
                    for (int k = 0; k < NHP_NT; ++k) {
                        float v = 0.f;
                        for (int w8 = 0; w8 < NWG; ++w8)
                            v += psum[((((long)buf * 6 + 1 + s) * NWG + w8) * NHP_B + b) * NHP_NT + k];
                        tot += splus(v);
                    }
                accI[b] += (tot * (1.f / NHP_S)) * dt;
            }
        }
    }

    if (wg == 0) {
        __syncthreads();
        if (threadIdx.x == 0) {
            float sI = 0.f, sL = 0.f;
            for (int b = 0; b < NHP_B; ++b) { sI += accI[b]; sL += accL[b]; }
            out[0] = sI - sL;
        }
    }
}

extern "C" void kernel_launch(void* const* d_in, const int* in_sizes, int n_in,
                              void* d_out, int out_size, void* d_ws, size_t ws_size,
                              hipStream_t stream) {
    const int*   type_seq = (const int*)  d_in[0];
    const float* time_seq = (const float*)d_in[1];
    const float* uu       = (const float*)d_in[2];
    const float* embed    = (const float*)d_in[3];
    const float* W_gates  = (const float*)d_in[4];
    const float* b_gates  = (const float*)d_in[5];
    const float* W_int    = (const float*)d_in[6];
    float* out = (float*)d_out;

    char* ws = (char*)d_ws;
    unsigned short* Xts = (unsigned short*)(ws + WS_X);
    unsigned short* Hts = (unsigned short*)(ws + WS_H);
    float*          ps  = (float*)(ws + WS_PSUM);
    unsigned*       bar = (unsigned*)(ws + WS_BAR);

    (void)hipFuncSetAttribute((const void*)nhp_scan,
                              hipFuncAttributeMaxDynamicSharedMemorySize, LDS_BYTES);

    long xElems = (long)NHP_L * 16 * 512;
    int pgrid = (int)((xElems + 255) / 256);
    nhp_prologue<<<pgrid, 256, 0, stream>>>(type_seq, embed, Xts, Hts, bar);
    nhp_scan<<<NWG, 256, LDS_BYTES, stream>>>(type_seq, time_seq, uu, W_gates, b_gates,
                                              W_int, Xts, Hts, ps, bar, out);
}
